// FormulaNet_30090540876123
// MI455X (gfx1250) — matrix operand order repaired and checked by reading, not compiled, not run on hardware
//
#include <hip/hip_runtime.h>

// ---------------------------------------------------------------------------
// FormulaNet fully fused kernel for gfx1250 (MI455X).
// All weights + embedding tables live in LDS (f16, transposed, zero-padded to
// WMMA tiles). Each block processes tiles of 16 binary nodes (= 32 predicate
// nodes); the entire pred->bin->un->univ->fin chain runs in LDS, only the
// final [M,155] fp32 output is written to HBM.
// Matrix cores: v_wmma_f32_16x16x32_f16 (f16 in, f32 accumulate).
// CDNA5 data movement: GLOBAL_LOAD_ASYNC_TO_LDS (indices, ASYNCcnt) and
// TENSOR_LOAD_TO_LDS (TDM weight staging, TENSORcnt), both guarded so the
// kernel still builds if a toolchain lacks the builtins.
// ---------------------------------------------------------------------------

typedef _Float16 h16;
typedef __attribute__((ext_vector_type(16))) _Float16 v16h;
typedef __attribute__((ext_vector_type(8)))  _Float16 v8h;
typedef __attribute__((ext_vector_type(8)))  float    v8f;

#define FN_L     50      // feature length
#define FN_NP    64      // padded col count for 50-wide hidden layers
#define FN_KPA   320     // padded K for pred matmul (6*50=300 -> 320)
#define FN_KPB   160     // padded K for bin matmul (3*50=150 -> 160)
#define FN_OUTC  155     // output cols
#define FN_OUTP  160     // padded output cols
#define FN_VOCAB 100

// ---- CDNA5 feature gates ---------------------------------------------------
#if __has_builtin(__builtin_amdgcn_global_load_async_to_lds_b128) && \
    __has_builtin(__builtin_amdgcn_s_wait_asynccnt)
#define FN_ASYNC 1
#else
#define FN_ASYNC 0
#endif

#if __has_builtin(__builtin_amdgcn_tensor_load_to_lds) && \
    __has_builtin(__builtin_amdgcn_s_wait_tensorcnt)
#define FN_TDM 1
#else
#define FN_TDM 0
#endif

// Builtin prototype (from hipcc diagnostic): vV4i*1 V4i*3 Ii Ii — pointers to
// GCC-style int __vector(4) in AS1/AS3. Synthesize via exact-width integer
// casts (always compile-legal; AS3 pointers are 32-bit).
typedef int fn_v4i __attribute__((vector_size(16)));
#define FN_GLB4(p) ((__attribute__((address_space(1))) fn_v4i*) \
                        (unsigned long long)(p))
#define FN_LDS4(p) ((__attribute__((address_space(3))) fn_v4i*) \
                        (unsigned)(unsigned long long)(p))

// ----- dynamic LDS layout (byte offsets / h16-element offsets) --------------
constexpr int SM_PID  = 0;     // 32  ints
constexpr int SM_VID  = 128;   // 160 ints
constexpr int SM_OID  = 768;   // 16  ints
constexpr int SM_BP   = 832;   // 64  f32
constexpr int SM_BB   = 1088;  // 64  f32
constexpr int SM_BU   = 1344;  // 64  f32
constexpr int SM_BV   = 1600;  // 64  f32
constexpr int SM_BF   = 1856;  // 160 f32
constexpr int SM_WATT = 2496;  // 64  f32
constexpr int SM_BATT = 2752;  // 1   f32
constexpr int SM_H    = 4096;  // base of f16 region (128B aligned)

constexpr int H_PE    = 0;                     // 100x64
constexpr int H_VE    = H_PE    + FN_VOCAB * 64;
constexpr int H_OE    = H_VE    + FN_VOCAB * 64;
constexpr int H_WTP   = H_OE    + FN_VOCAB * 64;       // [64][320]  W_pred^T
constexpr int H_WTB   = H_WTP   + FN_NP * FN_KPA;      // [64][160]  W_bin^T
constexpr int H_WTU   = H_WTB   + FN_NP * FN_KPB;      // [64][64]   W_un^T
constexpr int H_WTV   = H_WTU   + FN_NP * FN_NP;       // [64][64]   W_univ^T
constexpr int H_WTF   = H_WTV   + FN_NP * FN_NP;       // [160][64]  W_fin^T
constexpr int H_ARGS  = H_WTF   + FN_OUTP * FN_NP;     // [32][320] (also TDM scratch)
constexpr int H_HPRED = H_ARGS  + 32 * FN_KPA;         // [32][64]
constexpr int H_OPS   = H_HPRED + 32 * FN_NP;          // [16][160]
constexpr int H_HBIN  = H_OPS   + 16 * FN_KPB;         // [16][64]
constexpr int H_HUN   = H_HBIN  + 16 * FN_NP;          // [16][64]
constexpr int H_HQ    = H_HUN   + 16 * FN_NP;          // [16][64]
constexpr int H_TOTAL = H_HQ    + 16 * FN_NP;

constexpr size_t SMEM_BYTES = (size_t)SM_H + (size_t)H_TOTAL * 2;  // ~172.5 KB

// ----- WMMA fragment loaders (ISA 7.12.2 layouts, wave32) -------------------
__device__ __forceinline__ v16h ldA(const h16* feat, int KP, int m0, int k0,
                                    int lane) {
  int row = m0 + (lane & 15);
  int kb  = k0 + ((lane & 16) ? 8 : 0);
  const h16* p = feat + row * KP + kb;
  union { v16h v; v8h h[2]; } u;
  u.h[0] = *(const v8h*)(p);
  u.h[1] = *(const v8h*)(p + 16);
  return u.v;
}

__device__ __forceinline__ v16h ldB(const h16* wt, int KP, int n0, int k0,
                                    int lane) {
  int col = n0 + (lane & 15);
  int kb  = k0 + ((lane & 16) ? 8 : 0);
  const h16* p = wt + col * KP + kb;
  union { v16h v; v8h h[2]; } u;
  u.h[0] = *(const v8h*)(p);
  u.h[1] = *(const v8h*)(p + 16);
  return u.v;
}

__device__ __forceinline__ void store_relu(h16* dst, const v8f& acc,
                                           const float* bias, int m0, int n0,
                                           int lane, int ncols) {
  int col   = n0 + (lane & 15);
  int rbase = m0 + ((lane & 16) ? 8 : 0);
#pragma unroll
  for (int r = 0; r < 8; ++r) {
    float v = 0.f;
    if (col < ncols) {
      v = acc[r] + bias[col];
      v = v > 0.f ? v : 0.f;
    }
    dst[(rbase + r) * FN_NP + col] = (h16)v;
  }
}

#if FN_TDM
typedef __attribute__((ext_vector_type(4))) unsigned int u32x4;
typedef __attribute__((ext_vector_type(8))) int          i32x8;
typedef __attribute__((ext_vector_type(4))) int          i32x4;

// TDM DMA of a contiguous run of fp32 elements (1-D tile) global -> LDS.
// D# packing per CDNA5 ISA ch.8: group0 {count=1, lds_addr, global_addr,
// type=2}; group1 {data_size=4B, tensor_dim0=tile_dim0=nelem, stride0=nelem}.
__device__ __forceinline__ void tdm_copy_f32(const float* gsrc, void* ldst,
                                             unsigned nelem) {
  unsigned long long ga = (unsigned long long)gsrc;
  unsigned la = (unsigned)(unsigned long long)ldst;
  u32x4 g0;
  g0[0] = 1u;                                          // count=1, user desc
  g0[1] = la;                                          // lds_addr (bytes)
  g0[2] = (unsigned)(ga & 0xffffffffu);                // global_addr[31:0]
  g0[3] = (unsigned)((ga >> 32) & 0x01ffffffu)         // global_addr[56:32]
          | (2u << 30);                                // type = 2 ("image")
  i32x8 g1;
  g1[0] = (int)(2u << 16);                             // data_size = 4B
  g1[1] = (int)((nelem & 0xffffu) << 16);              // tensor_dim0[15:0]
  g1[2] = (int)((nelem >> 16) & 0xffffu);              // tensor_dim0[31:16]
  g1[3] = (int)((nelem & 0xffffu) << 16);              // tile_dim0
  g1[4] = 0;                                           // tile_dim1/2 unused
  g1[5] = (int)nelem;                                  // tensor_dim0_stride
  g1[6] = 0;
  g1[7] = 0;
  i32x4 gz = {0, 0, 0, 0};
#if __clang_major__ >= 23
  i32x8 gz8 = {0, 0, 0, 0, 0, 0, 0, 0};
  __builtin_amdgcn_tensor_load_to_lds(g0, g1, gz, gz, gz8, 0);
#else
  __builtin_amdgcn_tensor_load_to_lds(g0, g1, gz, gz, 0);
#endif
}
#endif  // FN_TDM

__global__ __launch_bounds__(256)
void formulanet_fused(const float* __restrict__ g_pe, const float* __restrict__ g_ve,
                      const float* __restrict__ g_oe,
                      const float* __restrict__ Wp, const float* __restrict__ bp,
                      const float* __restrict__ Wb, const float* __restrict__ bb,
                      const float* __restrict__ Wu, const float* __restrict__ bu,
                      const float* __restrict__ Wv, const float* __restrict__ bv,
                      const float* __restrict__ Wa, const float* __restrict__ ba,
                      const float* __restrict__ Wf, const float* __restrict__ bf,
                      const int* __restrict__ pred_ids,
                      const int* __restrict__ var_ids,
                      const int* __restrict__ op_ids,
                      float* __restrict__ out, int ntiles) {
  extern __shared__ char smem[];
  int*   s_pid  = (int*)(smem + SM_PID);
  int*   s_vid  = (int*)(smem + SM_VID);
  int*   s_oid  = (int*)(smem + SM_OID);
  float* s_bp   = (float*)(smem + SM_BP);
  float* s_bb   = (float*)(smem + SM_BB);
  float* s_bu   = (float*)(smem + SM_BU);
  float* s_bv   = (float*)(smem + SM_BV);
  float* s_bf   = (float*)(smem + SM_BF);
  float* s_watt = (float*)(smem + SM_WATT);
  float* s_batt = (float*)(smem + SM_BATT);
  h16*   s_h    = (h16*)(smem + SM_H);
  h16*   s_pe   = s_h + H_PE;
  h16*   s_ve   = s_h + H_VE;
  h16*   s_oe   = s_h + H_OE;
  h16*   s_wtp  = s_h + H_WTP;
  h16*   s_wtb  = s_h + H_WTB;
  h16*   s_wtu  = s_h + H_WTU;
  h16*   s_wtv  = s_h + H_WTV;
  h16*   s_wtf  = s_h + H_WTF;
  h16*   s_args = s_h + H_ARGS;
  h16*   s_hpred= s_h + H_HPRED;
  h16*   s_ops  = s_h + H_OPS;
  h16*   s_hbin = s_h + H_HBIN;
  h16*   s_hun  = s_h + H_HUN;
  h16*   s_hq   = s_h + H_HQ;

  const int tid  = threadIdx.x;
  const int lane = tid & 31;
  const int wave = tid >> 5;

  // ---- one-time per-block weight staging (fp32 -> f16, transposed+padded) --
#if FN_TDM
  // Kick TDM DMA of raw W_un into the (idle) s_args scratch before the big
  // staging loops; DMA overlaps the VALU/LDS conversion work below.
  if (wave == 0) tdm_copy_f32(Wu, s_args, FN_L * FN_L);
#endif
  for (int e = tid; e < FN_VOCAB * 64; e += 256) {
    int r = e >> 6, c = e & 63;
    float v0 = (c < FN_L) ? g_pe[r * FN_L + c] : 0.f;
    float v1 = (c < FN_L) ? g_ve[r * FN_L + c] : 0.f;
    float v2 = (c < FN_L) ? g_oe[r * FN_L + c] : 0.f;
    s_pe[e] = (h16)v0; s_ve[e] = (h16)v1; s_oe[e] = (h16)v2;
  }
  for (int e = tid; e < FN_NP * FN_KPA; e += 256) {
    int n = e / FN_KPA, k = e % FN_KPA;
    s_wtp[e] = (h16)((n < FN_L && k < 6 * FN_L) ? Wp[k * FN_L + n] : 0.f);
  }
  for (int e = tid; e < FN_NP * FN_KPB; e += 256) {
    int n = e / FN_KPB, k = e % FN_KPB;
    s_wtb[e] = (h16)((n < FN_L && k < 3 * FN_L) ? Wb[k * FN_L + n] : 0.f);
  }
  for (int e = tid; e < FN_OUTP * FN_NP; e += 256) {
    int n = e / FN_NP, k = e % FN_NP;
    s_wtf[e] = (h16)((n < FN_OUTC && k < FN_L) ? Wf[k * FN_OUTC + n] : 0.f);
  }
  if (tid < 64) {
    s_bp[tid]   = (tid < FN_L) ? bp[tid] : 0.f;
    s_bb[tid]   = (tid < FN_L) ? bb[tid] : 0.f;
    s_bu[tid]   = (tid < FN_L) ? bu[tid] : 0.f;
    s_bv[tid]   = (tid < FN_L) ? bv[tid] : 0.f;
    s_watt[tid] = (tid < FN_L) ? Wa[tid] : 0.f;
  }
  if (tid < FN_OUTP) s_bf[tid] = (tid < FN_OUTC) ? bf[tid] : 0.f;
  if (tid == 0) s_batt[0] = ba[0];

#if FN_TDM
  // W_un: TDM landed raw fp32 in s_args -> convert to transposed f16 tiles.
  __builtin_amdgcn_s_wait_tensorcnt(0);   // only wave 0 has TENSORcnt pending
  __syncthreads();
  {
    const float* raw = (const float*)(const void*)s_args;
    for (int e = tid; e < FN_NP * FN_NP; e += 256) {
      int n = e / FN_NP, k = e % FN_NP;
      s_wtu[e] = (h16)((n < FN_L && k < FN_L) ? raw[k * FN_L + n] : 0.f);
    }
  }
  __syncthreads();
  if (wave == 0) tdm_copy_f32(Wv, s_args, FN_L * FN_L);
  __builtin_amdgcn_s_wait_tensorcnt(0);
  __syncthreads();
  {
    const float* raw = (const float*)(const void*)s_args;
    for (int e = tid; e < FN_NP * FN_NP; e += 256) {
      int n = e / FN_NP, k = e % FN_NP;
      s_wtv[e] = (h16)((n < FN_L && k < FN_L) ? raw[k * FN_L + n] : 0.f);
    }
  }
#else
  for (int e = tid; e < FN_NP * FN_NP; e += 256) {
    int n = e / FN_NP, k = e % FN_NP;
    bool in = (n < FN_L && k < FN_L);
    s_wtu[e] = (h16)(in ? Wu[k * FN_L + n] : 0.f);
    s_wtv[e] = (h16)(in ? Wv[k * FN_L + n] : 0.f);
  }
#endif
  __syncthreads();

  // ---- grid-stride loop over tiles of 16 bin nodes (= 32 pred nodes) -------
  for (int tile = blockIdx.x; tile < ntiles; tile += gridDim.x) {
    const int pbase = tile * 32;
    const int obase = tile * 16;

    // prefetch next tile's indices (global_prefetch_b8)
    int nxt = tile + gridDim.x;
    if (nxt < ntiles && tid == 0) {
      __builtin_prefetch(pred_ids + nxt * 32, 0, 0);
      __builtin_prefetch(var_ids + nxt * 160, 0, 0);
      __builtin_prefetch(op_ids + nxt * 16, 0, 0);
    }

#if FN_ASYNC
    // DMA the tile's indices straight into LDS (ASYNCcnt path): 16B per lane.
    if (tid < 8)
      __builtin_amdgcn_global_load_async_to_lds_b128(
          FN_GLB4(pred_ids + pbase + tid * 4), FN_LDS4(s_pid + tid * 4), 0, 0);
    if (tid < 40)
      __builtin_amdgcn_global_load_async_to_lds_b128(
          FN_GLB4(var_ids + pbase * 5 + tid * 4), FN_LDS4(s_vid + tid * 4), 0, 0);
    if (tid < 4)
      __builtin_amdgcn_global_load_async_to_lds_b128(
          FN_GLB4(op_ids + obase + tid * 4), FN_LDS4(s_oid + tid * 4), 0, 0);
    __builtin_amdgcn_s_wait_asynccnt(0);
#else
    if (tid < 32)  s_pid[tid] = pred_ids[pbase + tid];
    if (tid < 160) s_vid[tid] = var_ids[pbase * 5 + tid];
    if (tid < 16)  s_oid[tid] = op_ids[obase + tid];
#endif
    __syncthreads();

    // gather args tile [32][320] from LDS-resident embedding tables
    for (int e = tid; e < 32 * FN_KPA; e += 256) {
      int r = e / FN_KPA, k = e % FN_KPA;
      h16 v = (h16)0.f;
      if (k < 6 * FN_L) {
        int seg = k / FN_L, kk = k - seg * FN_L;
        if (seg == 0) v = s_pe[s_pid[r] * 64 + kk];
        else          v = s_ve[s_vid[r * 5 + seg - 1] * 64 + kk];
      }
      s_args[e] = v;
    }
    __syncthreads();

    // ---- pred GEMM: [32 x 320] x [320 x 64], 8 wave jobs -------------------
    {
      const int m0 = (wave >> 2) * 16, n0 = (wave & 3) * 16;
      v8f acc = {};
#pragma unroll
      for (int kt = 0; kt < FN_KPA / 32; ++kt)
        acc = __builtin_amdgcn_wmma_f32_16x16x32_f16(
            false, ldA(s_args, FN_KPA, m0, kt * 32, lane),
            false, ldB(s_wtp, FN_KPA, n0, kt * 32, lane),
            (short)0, acc, false, false);
      store_relu(s_hpred, acc, s_bp, m0, n0, lane, FN_L);
    }
    __syncthreads();

    // ---- pred attention: softmax over 6 arg segments, one row per lane -----
    if (tid < 32) {
      const h16* row = s_args + tid * FN_KPA;
      float lg[6];
#pragma unroll
      for (int j = 0; j < 6; ++j) {
        float s = s_batt[0];
        for (int k = 0; k < FN_L; ++k)
          s += (float)row[j * FN_L + k] * s_watt[k];
        lg[j] = s;
      }
      float mx = lg[0];
#pragma unroll
      for (int j = 1; j < 6; ++j) mx = fmaxf(mx, lg[j]);
      float w[6], den = 0.f;
#pragma unroll
      for (int j = 0; j < 6; ++j) { w[j] = __expf(lg[j] - mx); den += w[j]; }
      float inv = 1.f / den;
      for (int k = 0; k < FN_L; ++k) {
        float a = 0.f;
#pragma unroll
        for (int j = 0; j < 6; ++j) a += w[j] * (float)row[j * FN_L + k];
        h16* p = s_hpred + tid * FN_NP + k;
        *p = (h16)((float)*p + a * inv);
      }
    }
    __syncthreads();

    // ---- build ops tile [16][160] = [left | op_emb | right] ----------------
    for (int e = tid; e < 16 * FN_KPB; e += 256) {
      int r = e / FN_KPB, k = e % FN_KPB;
      h16 v = (h16)0.f;
      if (k < FN_L)            v = s_hpred[(2 * r) * FN_NP + k];
      else if (k < 2 * FN_L)   v = s_oe[s_oid[r] * 64 + (k - FN_L)];
      else if (k < 3 * FN_L)   v = s_hpred[(2 * r + 1) * FN_NP + (k - 2 * FN_L)];
      s_ops[e] = v;
    }
    __syncthreads();

    // ---- bin GEMM: [16 x 160] x [160 x 64], 4 wave jobs --------------------
    if (wave < 4) {
      const int n0 = wave * 16;
      v8f acc = {};
#pragma unroll
      for (int kt = 0; kt < FN_KPB / 32; ++kt)
        acc = __builtin_amdgcn_wmma_f32_16x16x32_f16(
            false, ldA(s_ops, FN_KPB, 0, kt * 32, lane),
            false, ldB(s_wtb, FN_KPB, n0, kt * 32, lane),
            (short)0, acc, false, false);
      store_relu(s_hbin, acc, s_bb, 0, n0, lane, FN_L);
    }
    __syncthreads();

    // ---- bin attention: softmax over 3 segments ----------------------------
    if (tid < 16) {
      const h16* row = s_ops + tid * FN_KPB;
      float lg[3];
#pragma unroll
      for (int j = 0; j < 3; ++j) {
        float s = s_batt[0];
        for (int k = 0; k < FN_L; ++k)
          s += (float)row[j * FN_L + k] * s_watt[k];
        lg[j] = s;
      }
      float mx = fmaxf(lg[0], fmaxf(lg[1], lg[2]));
      float w[3], den = 0.f;
#pragma unroll
      for (int j = 0; j < 3; ++j) { w[j] = __expf(lg[j] - mx); den += w[j]; }
      float inv = 1.f / den;
      for (int k = 0; k < FN_L; ++k) {
        float a = 0.f;
#pragma unroll
        for (int j = 0; j < 3; ++j) a += w[j] * (float)row[j * FN_L + k];
        h16* p = s_hbin + tid * FN_NP + k;
        *p = (h16)((float)*p + a * inv);
      }
    }
    __syncthreads();

    // ---- un: relu(h_bin @ W_un + b_un) -------------------------------------
    if (wave < 4) {
      const int n0 = wave * 16;
      v8f acc = {};
#pragma unroll
      for (int kt = 0; kt < 2; ++kt)
        acc = __builtin_amdgcn_wmma_f32_16x16x32_f16(
            false, ldA(s_hbin, FN_NP, 0, kt * 32, lane),
            false, ldB(s_wtu, FN_NP, n0, kt * 32, lane),
            (short)0, acc, false, false);
      store_relu(s_hun, acc, s_bu, 0, n0, lane, FN_L);
    }
    __syncthreads();

    // ---- univ: relu(h_un @ W_univ + b_univ) --------------------------------
    if (wave < 4) {
      const int n0 = wave * 16;
      v8f acc = {};
#pragma unroll
      for (int kt = 0; kt < 2; ++kt)
        acc = __builtin_amdgcn_wmma_f32_16x16x32_f16(
            false, ldA(s_hun, FN_NP, 0, kt * 32, lane),
            false, ldB(s_wtv, FN_NP, n0, kt * 32, lane),
            (short)0, acc, false, false);
      store_relu(s_hq, acc, s_bv, 0, n0, lane, FN_L);
    }
    __syncthreads();

    // ---- fin: h_q @ W_fin + b_fin -> global [16 x 155] ---------------------
    for (int j = wave; j < FN_OUTP / 16; j += 8) {
      const int n0 = j * 16;
      v8f acc = {};
#pragma unroll
      for (int kt = 0; kt < 2; ++kt)
        acc = __builtin_amdgcn_wmma_f32_16x16x32_f16(
            false, ldA(s_hq, FN_NP, 0, kt * 32, lane),
            false, ldB(s_wtf, FN_NP, n0, kt * 32, lane),
            (short)0, acc, false, false);
      int col = n0 + (lane & 15);
      if (col < FN_OUTC) {
        int rbase = obase + ((lane & 16) ? 8 : 0);
#pragma unroll
        for (int r = 0; r < 8; ++r)
          out[(rbase + r) * FN_OUTC + col] = acc[r] + s_bf[col];
      }
    }
    __syncthreads();
  }
}

extern "C" void kernel_launch(void* const* d_in, const int* in_sizes, int n_in,
                              void* d_out, int out_size, void* d_ws, size_t ws_size,
                              hipStream_t stream) {
  const float* pe = (const float*)d_in[0];
  const float* ve = (const float*)d_in[1];
  const float* oe = (const float*)d_in[2];
  const float* Wp = (const float*)d_in[3];
  const float* bp = (const float*)d_in[4];
  const float* Wb = (const float*)d_in[5];
  const float* bb = (const float*)d_in[6];
  const float* Wu = (const float*)d_in[7];
  const float* bu = (const float*)d_in[8];
  const float* Wv = (const float*)d_in[9];
  const float* bv = (const float*)d_in[10];
  const float* Wa = (const float*)d_in[11];
  const float* ba = (const float*)d_in[12];
  const float* Wf = (const float*)d_in[13];
  const float* bf = (const float*)d_in[14];
  const int* pred_ids = (const int*)d_in[15];
  const int* var_ids  = (const int*)d_in[16];
  const int* op_ids   = (const int*)d_in[17];
  float* out = (float*)d_out;

  const int M = in_sizes[17];          // number of binary-formula nodes
  const int ntiles = M / 16;
  int grid = ntiles < 1024 ? ntiles : 1024;

  formulanet_fused<<<grid, 256, SMEM_BYTES, stream>>>(
      pe, ve, oe, Wp, bp, Wb, bb, Wu, bu, Wv, bv, Wa, ba, Wf, bf,
      pred_ids, var_ids, op_ids, out, ntiles);
}
